// GNN_encoder_86878598464218
// MI455X (gfx1250) — matrix operand order repaired and checked by reading, not compiled, not run on hardware
//
#include <hip/hip_runtime.h>
#include <hip/hip_bf16.h>

#define N_NODES   40000
#define N_PAD     40064      // 313 * 128
#define E_EDGES   640000
#define D_DIM     128
#define H_DIM     256
#define L_TOK     10
#define NL_LAYERS 5
#define NE_TYPES  10
#define G_GRAPHS  512
#define BN_EPS    1e-5f

typedef __attribute__((ext_vector_type(16))) __bf16 v16bf;
typedef __attribute__((ext_vector_type(8)))  __bf16 v8bf;
typedef __attribute__((ext_vector_type(8)))  float  v8f;

static __device__ __forceinline__ unsigned short f2bf(float f) {
    unsigned int u = __float_as_uint(f);
    unsigned int r = (u + 0x7FFFu + ((u >> 16) & 1u)) >> 16;
    return (unsigned short)r;
}
static __device__ __forceinline__ float bf2f(unsigned short h) {
    return __uint_as_float(((unsigned int)h) << 16);
}

// ---------------- WordBag: mean of valid subtoken embeddings -> hb (bf16) ----------------
__global__ void k_wordbag(const int* __restrict__ xt, const int* __restrict__ len,
                          const float* __restrict__ wemb, unsigned short* __restrict__ hb) {
    const int n = blockIdx.x;
    const int c = threadIdx.x;
    const int ln = len[n];
    float s = 0.f;
    for (int l = 0; l < ln; ++l) {
        const int t = xt[n * L_TOK + l];
        s += wemb[(size_t)t * D_DIM + c];
    }
    hb[(size_t)n * D_DIM + c] = f2bf(s / (float)ln);
}

// --- Transpose + convert weights to bf16 (B fragments contiguous) + bf16 edge embeddings ---
__global__ void k_prep_weights(const float* __restrict__ W1, const float* __restrict__ W2,
                               const float* __restrict__ eemb,
                               unsigned short* __restrict__ W1T, unsigned short* __restrict__ W2T,
                               unsigned short* __restrict__ eb) {
    const int i = blockIdx.x * blockDim.x + threadIdx.x;     // [0, NL*D*H)
    if (i >= NL_LAYERS * D_DIM * H_DIM) return;
    if (i < NE_TYPES * D_DIM) eb[i] = f2bf(eemb[i]);
    const int l = i / (D_DIM * H_DIM);
    const int r = i % (D_DIM * H_DIM);
    // W1 [l][k(D)][n(H)] -> W1T [l][n(H)][k(D)]
    const int k1 = r / H_DIM, n1 = r % H_DIM;
    W1T[(size_t)l * H_DIM * D_DIM + (size_t)n1 * D_DIM + k1] = f2bf(W1[i]);
    // W2 [l][k(H)][n(D)] -> W2T [l][n(D)][k(H)]
    const int k2 = r / D_DIM, n2 = r % D_DIM;
    W2T[(size_t)l * D_DIM * H_DIM + (size_t)n2 * H_DIM + k2] = f2bf(W2[i]);
}

// ---- Edge message + scatter-add in bf16: aggrb[dst] += hb[src] + eb[type] ----
// thread = (edge, 8 features); 64 pk-bf16 atomics per edge instead of 128 f32 atomics.
__global__ void k_scatter(const int* __restrict__ ei, const int* __restrict__ ea,
                          const unsigned short* __restrict__ eb,
                          const unsigned short* __restrict__ hb,
                          unsigned short* __restrict__ aggrb) {
    const long idx = (long)blockIdx.x * blockDim.x + threadIdx.x;
    const int e = (int)(idx >> 4);
    const int f = ((int)idx & 15) * 8;
    if (e >= E_EDGES) return;
    const int src = ei[e], dst = ei[E_EDGES + e], t = ea[e];
    union U8 { v8bf v; uint4 q; unsigned int w[4]; };
    U8 hv, ev;
    hv.q = *(const uint4*)(hb + (size_t)src * D_DIM + f);
    ev.q = *(const uint4*)(eb + (size_t)t * D_DIM + f);
    hv.v = hv.v + ev.v;                                   // v_pk_add_bf16
    const unsigned long long base =
        (unsigned long long)(aggrb + (size_t)dst * D_DIM + f);
    #pragma unroll
    for (int j = 0; j < 4; ++j) {
        asm volatile("global_atomic_pk_add_bf16 %0, %1, off"
                     :: "v"(base + 4ull * (unsigned long long)j), "v"(hv.w[j])
                     : "memory");
    }
}

// ------------- Fused GIN MLP: z = relu((hb+aggrb)@W1 + b1)@W2 + b2 (WMMA bf16) -------------
// block = 256 threads = 8 waves; each wave owns a 16-row strip (block tile: 128 rows)
__global__ void __launch_bounds__(256) k_gin_mlp(
        const unsigned short* __restrict__ hb,
        const unsigned short* __restrict__ aggrb,
        const unsigned short* __restrict__ W1T,   // [H][D] bf16
        const unsigned short* __restrict__ W2T,   // [D][H] bf16
        const float* __restrict__ b1, const float* __restrict__ b2,
        unsigned short* __restrict__ zoutb, float* __restrict__ gsum, float* __restrict__ gsq) {
    __shared__ unsigned short inter[8][16][H_DIM];   // 64 KB: per-wave 16x256 bf16 staging
    const int wave = threadIdx.x >> 5;
    const int lane = threadIdx.x & 31;
    const int half = lane >> 4;     // K-half selector for A/B fragments
    const int l16  = lane & 15;
    const int rowTile = blockIdx.x * 128 + wave * 16;

    union frag { v16bf v; uint4 q[2]; };
    union U8 { v8bf v; uint4 q; };

    // ---- Build A fragments: a = bf16(h + aggr), fused (no separate pack pass) ----
    frag a1[4];
    {
        const size_t rowOff = (size_t)(rowTile + l16) * D_DIM + half * 8;
        const unsigned short* hp = hb + rowOff;
        const unsigned short* gp = aggrb + rowOff;
        #pragma unroll
        for (int kt = 0; kt < 4; ++kt) {
            U8 x0, y0, x1, y1;
            x0.q = *(const uint4*)(hp + kt * 32);
            y0.q = *(const uint4*)(gp + kt * 32);
            x1.q = *(const uint4*)(hp + kt * 32 + 16);
            y1.q = *(const uint4*)(gp + kt * 32 + 16);
            x0.v = x0.v + y0.v;                           // v_pk_add_bf16
            x1.v = x1.v + y1.v;
            a1[kt].q[0] = x0.q;
            a1[kt].q[1] = x1.q;
        }
    }

    // ---- GEMM1: [16x128]x[128x256], relu -> LDS bf16 ----
    #pragma unroll
    for (int nt = 0; nt < 16; ++nt) {
        v8f acc = {};
        const int n = nt * 16 + l16;
        const unsigned short* bp = W1T + (size_t)n * D_DIM + half * 8;
        #pragma unroll
        for (int kt = 0; kt < 4; ++kt) {
            frag bf;
            bf.q[0] = *(const uint4*)(bp + kt * 32);
            bf.q[1] = *(const uint4*)(bp + kt * 32 + 16);
            acc = __builtin_amdgcn_wmma_f32_16x16x32_bf16(false, a1[kt].v, false, bf.v,
                                                          (short)0, acc, false, false);
        }
        const float bias = b1[n];
        #pragma unroll
        for (int v = 0; v < 8; ++v) {
            float z = acc[v] + bias;
            z = z > 0.f ? z : 0.f;
            inter[wave][half * 8 + v][n] = f2bf(z);       // C layout: m = half*8+v
        }
    }

    // ---- GEMM2: [16x256]x[256x128] from LDS; epilogue + BN stats ----
    #pragma unroll
    for (int nt = 0; nt < 8; ++nt) {
        v8f acc = {};
        const int n = nt * 16 + l16;
        const unsigned short* bp  = W2T + (size_t)n * H_DIM + half * 8;
        const unsigned short* ap2 = &inter[wave][l16][half * 8];
        #pragma unroll
        for (int kt = 0; kt < 8; ++kt) {
            frag af, bf;
            af.q[0] = *(const uint4*)(ap2 + kt * 32);
            af.q[1] = *(const uint4*)(ap2 + kt * 32 + 16);
            bf.q[0] = *(const uint4*)(bp + kt * 32);
            bf.q[1] = *(const uint4*)(bp + kt * 32 + 16);
            acc = __builtin_amdgcn_wmma_f32_16x16x32_bf16(false, af.v, false, bf.v,
                                                          (short)0, acc, false, false);
        }
        const float bias = b2[n];
        float ps = 0.f, pq = 0.f;
        #pragma unroll
        for (int v = 0; v < 8; ++v) {
            const int row = rowTile + half * 8 + v;
            const float z = acc[v] + bias;
            if (row < N_NODES) {
                zoutb[(size_t)row * D_DIM + n] = f2bf(z);
                ps += z;
                pq += z * z;
            }
        }
        atomicAdd(&gsum[n], ps);    // lane-folded: 2 atomics per (lane, n-tile)
        atomicAdd(&gsq[n],  pq);
    }
}

// ------------- BatchNorm (training stats) + optional ReLU -------------
// mode 0: write bf16 h for next layer (with ReLU); mode 1: write f32 node_rep (no ReLU)
__global__ void k_bn_apply(const unsigned short* __restrict__ zoutb,
                           const float* __restrict__ gsum, const float* __restrict__ gsq,
                           const float* __restrict__ gamma, const float* __restrict__ beta,
                           unsigned short* __restrict__ hb_out, float* __restrict__ f32_out,
                           int mode) {
    const long i = (long)blockIdx.x * blockDim.x + threadIdx.x;   // exactly N*D threads
    const int c = (int)i & 127;
    const float invN = 1.f / (float)N_NODES;
    const float mu  = gsum[c] * invN;
    const float var = gsq[c] * invN - mu * mu;
    float z = (bf2f(zoutb[i]) - mu) * rsqrtf(var + BN_EPS) * gamma[c] + beta[c];
    if (mode == 0) {
        z = z > 0.f ? z : 0.f;
        hb_out[i] = f2bf(z);
    } else {
        f32_out[i] = z;
    }
}

// ------------- global mean pool -------------
__global__ void k_pool(const float* __restrict__ noderep, const int* __restrict__ bids,
                       float* __restrict__ gpool, float* __restrict__ gcnt) {
    const int n = blockIdx.x;
    const int c = threadIdx.x;
    const int b = bids[n];
    atomicAdd(&gpool[(size_t)b * D_DIM + c], noderep[(size_t)n * D_DIM + c]);
    if (c == 0) atomicAdd(&gcnt[b], 1.f);
}

__global__ void k_pool_fin(const float* __restrict__ gpool, const float* __restrict__ gcnt,
                           float* __restrict__ grep) {
    const int i = blockIdx.x * blockDim.x + threadIdx.x;          // G*D threads
    const int b = i >> 7;
    float cnt = gcnt[b];
    cnt = cnt > 1.f ? cnt : 1.f;
    grep[i] = gpool[i] / cnt;
}

extern "C" void kernel_launch(void* const* d_in, const int* in_sizes, int n_in,
                              void* d_out, int out_size, void* d_ws, size_t ws_size,
                              hipStream_t stream) {
    (void)in_sizes; (void)n_in; (void)out_size; (void)ws_size;
    const int*   x_tokens   = (const int*)d_in[0];
    const int*   ins_len    = (const int*)d_in[1];
    const int*   edge_index = (const int*)d_in[2];
    const int*   edge_attr  = (const int*)d_in[3];
    const int*   batch_ids  = (const int*)d_in[4];
    const float* word_emb   = (const float*)d_in[5];
    const float* edge_emb   = (const float*)d_in[6];
    const float* W1    = (const float*)d_in[7];
    const float* b1    = (const float*)d_in[8];
    const float* W2    = (const float*)d_in[9];
    const float* b2    = (const float*)d_in[10];
    const float* gamma = (const float*)d_in[11];
    const float* beta  = (const float*)d_in[12];

    char* ws = (char*)d_ws;
    size_t off = 0;
    auto carve = [&](size_t bytes) -> void* {
        void* p = ws + off;
        off = (off + bytes + 255) & ~(size_t)255;
        return p;
    };
    unsigned short* hb    = (unsigned short*)carve((size_t)N_PAD * D_DIM * 2);
    unsigned short* aggrb = (unsigned short*)carve((size_t)N_PAD * D_DIM * 2);
    unsigned short* zoutb = (unsigned short*)carve((size_t)N_PAD * D_DIM * 2);
    unsigned short* W1T   = (unsigned short*)carve((size_t)NL_LAYERS * D_DIM * H_DIM * 2);
    unsigned short* W2T   = (unsigned short*)carve((size_t)NL_LAYERS * D_DIM * H_DIM * 2);
    unsigned short* eb    = (unsigned short*)carve((size_t)NE_TYPES * D_DIM * 2);
    float*          gsum  = (float*)carve(D_DIM * 4);
    float*          gsq   = (float*)carve(D_DIM * 4);
    float*          gpool = (float*)carve((size_t)G_GRAPHS * D_DIM * 4);
    float*          gcnt  = (float*)carve(G_GRAPHS * 4);

    float* graph_out = (float*)d_out;
    float* node_out  = (float*)d_out + (size_t)G_GRAPHS * D_DIM;

    // zero the bf16 pad rows of hb once per launch (rows >= N feed WMMA as zeros)
    hipMemsetAsync(hb + (size_t)N_NODES * D_DIM, 0,
                   (size_t)(N_PAD - N_NODES) * D_DIM * 2, stream);

    k_wordbag<<<dim3(N_NODES), dim3(D_DIM), 0, stream>>>(x_tokens, ins_len, word_emb, hb);
    k_prep_weights<<<dim3((NL_LAYERS * D_DIM * H_DIM + 255) / 256), dim3(256), 0, stream>>>(
        W1, W2, edge_emb, W1T, W2T, eb);

    for (int l = 0; l < NL_LAYERS; ++l) {
        hipMemsetAsync(aggrb, 0, (size_t)N_PAD * D_DIM * 2, stream);
        hipMemsetAsync(gsum, 0, D_DIM * 4, stream);
        hipMemsetAsync(gsq,  0, D_DIM * 4, stream);
        k_scatter<<<dim3((E_EDGES * 16) / 256), dim3(256), 0, stream>>>(
            edge_index, edge_attr, eb, hb, aggrb);
        k_gin_mlp<<<dim3(N_PAD / 128), dim3(256), 0, stream>>>(
            hb, aggrb,
            W1T + (size_t)l * H_DIM * D_DIM,
            W2T + (size_t)l * D_DIM * H_DIM,
            b1 + l * H_DIM, b2 + l * D_DIM,
            zoutb, gsum, gsq);
        const int mode = (l == NL_LAYERS - 1) ? 1 : 0;
        k_bn_apply<<<dim3((N_NODES * D_DIM) / 256), dim3(256), 0, stream>>>(
            zoutb, gsum, gsq, gamma + l * D_DIM, beta + l * D_DIM, hb, node_out, mode);
    }

    hipMemsetAsync(gpool, 0, (size_t)G_GRAPHS * D_DIM * 4, stream);
    hipMemsetAsync(gcnt,  0, G_GRAPHS * 4, stream);
    k_pool<<<dim3(N_NODES), dim3(D_DIM), 0, stream>>>(node_out, batch_ids, gpool, gcnt);
    k_pool_fin<<<dim3((G_GRAPHS * D_DIM) / 256), dim3(256), 0, stream>>>(gpool, gcnt, graph_out);
}
